// ImprovedRecommenderModel_16157666967605
// MI455X (gfx1250) — matrix operand order repaired and checked by reading, not compiled, not run on hardware
//
#include <hip/hip_runtime.h>
#include <hip/hip_bf16.h>

typedef __attribute__((ext_vector_type(16))) _Float16 v16h;
typedef __attribute__((ext_vector_type(8)))  float    v8f;

#define B_TOTAL   262144
#define N_ISBNS   100000
#define SBERT_DIM 384
#define NUM_CATS  64
#define EMB       48
#define H_CAT     128
#define H_TXT     256
#define ALPHA_F   3.0f
#define BETA_F    1.0f
#define INV_TEMP  20.0f      // 1 / 0.05

// ---------------- WMMA helpers (gfx1250, wave32) ----------------

__device__ __forceinline__ v8f wmma_f16(v16h a, v16h b, v8f c) {
  // D = A(16x32 f16) * B(32x16 f16) + C(16x16 f32)
  return __builtin_amdgcn_wmma_f32_16x16x32_f16(
      /*neg_a=*/false, a, /*neg_b=*/false, b,
      /*c_mod=*/(short)0, c, /*reuse_a=*/false, /*reuse_b=*/false);
}

// A-matrix 16x32 f16 layout (ISA 7.12.2):
// lanes 0-15: M=lane, halves 0..7 -> K=0..7, halves 8..15 -> K=16..23
// lanes16-31: M=lane-16, halves 0..7 -> K=8..15, halves 8..15 -> K=24..31
__device__ __forceinline__ v16h load_a_frag_gf32(const float* __restrict__ base,
                                                 int ld, int lane) {
  const int laneM = lane & 15, laneHalf = lane >> 4;
  const float* rowp = base + (size_t)laneM * ld;
  v16h a;
#pragma unroll
  for (int h = 0; h < 16; ++h) {
    int k = laneHalf * 8 + ((h < 8) ? h : (h + 8));
    a[h] = (_Float16)rowp[k];
  }
  return a;
}

__device__ __forceinline__ v16h load_a_frag_lds(const _Float16* base, int ld, int lane) {
  const int laneM = lane & 15, laneHalf = lane >> 4;
  const _Float16* rowp = base + laneM * ld;
  v16h a;
#pragma unroll
  for (int h = 0; h < 16; ++h) {
    int k = laneHalf * 8 + ((h < 8) ? h : (h + 8));
    a[h] = rowp[k];
  }
  return a;
}

// B-matrix 32x16 f16: lanes 0-15 hold K=0..7,16..23, lanes 16-31 hold K=8..15,24..31,
// column N = lane%16.  Weights stored transposed [N][K] f16 -> two 8-half
// chunks per lane -> 2x global_load_b128.
__device__ __forceinline__ v16h load_b_frag(const _Float16* __restrict__ wt,
                                            int K, int ncol0, int kt, int lane) {
  const int laneM = lane & 15, laneHalf = lane >> 4;
  const _Float16* p = wt + (size_t)(ncol0 + laneM) * K + kt * 32 + laneHalf * 8;
  v16h b;
#pragma unroll
  for (int h = 0; h < 16; ++h) b[h] = p[(h < 8) ? h : (h + 8)];
  return b;
}

// ---------------- kernel: weight f32[K][N] -> f16[N][K] ----------------

__global__ void transpose_f16_kernel(const float* __restrict__ src,
                                     _Float16* __restrict__ dst, int K, int N) {
  int t = blockIdx.x * blockDim.x + threadIdx.x;
  if (t >= K * N) return;
  int k = t / N, n = t - k * N;
  dst[(size_t)n * K + k] = (_Float16)src[(size_t)k * N + n];
}

// ---------------- kernel: encode all 100000 books with WMMA ----------------

#define SH_HID_LD 264   // 256 + 8 halves pad (bank-conflict avoidance)
#define SH_OUT_LD 52

__global__ __launch_bounds__(128) void encode_books_kernel(
    const float* __restrict__ sbert,    // [N_ISBNS][384]
    const float* __restrict__ cat_mh,   // [N_ISBNS][64]
    const _Float16* __restrict__ Wc1T,  // [128][64]
    const float* __restrict__ bc1,      // [128]
    const _Float16* __restrict__ Wc2T,  // [48][128]
    const float* __restrict__ bc2,      // [48]
    const _Float16* __restrict__ Wt1T,  // [256][384]
    const float* __restrict__ bt1,      // [256]
    const _Float16* __restrict__ Wt2T,  // [48][256]
    const float* __restrict__ bt2,      // [48]
    float* __restrict__ bookemb)        // [N_ISBNS][48] l2-normalized
{
  __shared__ _Float16 sh_hid[4][16 * SH_HID_LD];
  __shared__ float    sh_out[4][16 * SH_OUT_LD];

  const int wave = threadIdx.x >> 5;
  const int lane = threadIdx.x & 31;
  const int laneM = lane & 15, laneHalf = lane >> 4;
  const int row0 = blockIdx.x * 64 + wave * 16;   // 16 books per wave
  if (row0 >= N_ISBNS) return;                    // wave-uniform

  _Float16* hid  = sh_hid[wave];
  float*    outb = sh_out[wave];

  // ================= CAT tower: relu(mh@Wc1+bc1)@Wc2+bc2 =================
  {
    v16h afr[NUM_CATS / 32];
#pragma unroll
    for (int kt = 0; kt < NUM_CATS / 32; ++kt)
      afr[kt] = load_a_frag_gf32(cat_mh + (size_t)row0 * NUM_CATS + kt * 32,
                                 NUM_CATS, lane);
    for (int nt = 0; nt < H_CAT / 16; ++nt) {
      v8f acc = {};
#pragma unroll
      for (int kt = 0; kt < NUM_CATS / 32; ++kt)
        acc = wmma_f16(afr[kt], load_b_frag(Wc1T, NUM_CATS, nt * 16, kt, lane), acc);
      int n = nt * 16 + laneM;
      float bias = bc1[n];
#pragma unroll
      for (int r = 0; r < 8; ++r) {
        int m = r + 8 * laneHalf;
        float v = acc[r] + bias;
        hid[m * SH_HID_LD + n] = (_Float16)(v > 0.f ? v : 0.f);
      }
    }
  }
  {
    v16h afr[H_CAT / 32];
#pragma unroll
    for (int kt = 0; kt < H_CAT / 32; ++kt)
      afr[kt] = load_a_frag_lds(hid + kt * 32, SH_HID_LD, lane);
#pragma unroll
    for (int nt = 0; nt < EMB / 16; ++nt) {
      v8f acc = {};
#pragma unroll
      for (int kt = 0; kt < H_CAT / 32; ++kt)
        acc = wmma_f16(afr[kt], load_b_frag(Wc2T, H_CAT, nt * 16, kt, lane), acc);
      int n = nt * 16 + laneM;
      float bias = bc2[n];
#pragma unroll
      for (int r = 0; r < 8; ++r) {
        int m = r + 8 * laneHalf;
        outb[m * SH_OUT_LD + n] = ALPHA_F * (acc[r] + bias);
      }
    }
  }

  // ================= TXT tower: relu(sbert@Wt1+bt1)@Wt2+bt2 =================
  {
    v16h afr[SBERT_DIM / 32];           // 12 fragments, reused across 16 N-tiles
#pragma unroll
    for (int kt = 0; kt < SBERT_DIM / 32; ++kt)
      afr[kt] = load_a_frag_gf32(sbert + (size_t)row0 * SBERT_DIM + kt * 32,
                                 SBERT_DIM, lane);
    for (int nt = 0; nt < H_TXT / 16; ++nt) {
      v8f acc = {};
#pragma unroll
      for (int kt = 0; kt < SBERT_DIM / 32; ++kt)
        acc = wmma_f16(afr[kt], load_b_frag(Wt1T, SBERT_DIM, nt * 16, kt, lane), acc);
      int n = nt * 16 + laneM;
      float bias = bt1[n];
#pragma unroll
      for (int r = 0; r < 8; ++r) {
        int m = r + 8 * laneHalf;
        float v = acc[r] + bias;
        hid[m * SH_HID_LD + n] = (_Float16)(v > 0.f ? v : 0.f);
      }
    }
  }
  {
    v16h afr[H_TXT / 32];
#pragma unroll
    for (int kt = 0; kt < H_TXT / 32; ++kt)
      afr[kt] = load_a_frag_lds(hid + kt * 32, SH_HID_LD, lane);
#pragma unroll
    for (int nt = 0; nt < EMB / 16; ++nt) {
      v8f acc = {};
#pragma unroll
      for (int kt = 0; kt < H_TXT / 32; ++kt)
        acc = wmma_f16(afr[kt], load_b_frag(Wt2T, H_TXT, nt * 16, kt, lane), acc);
      int n = nt * 16 + laneM;
      float bias = bt2[n];
#pragma unroll
      for (int r = 0; r < 8; ++r) {
        int m = r + 8 * laneHalf;
        outb[m * SH_OUT_LD + n] += BETA_F * (acc[r] + bias);
      }
    }
  }

  // ================= epilogue: per-row l2norm, store =================
  if (lane < 16) {
    int row = row0 + lane;
    if (row < N_ISBNS) {
      const float* rp = outb + lane * SH_OUT_LD;
      float s = 0.f;
#pragma unroll
      for (int j = 0; j < EMB; ++j) s += rp[j] * rp[j];
      float inv = rsqrtf(fmaxf(s, 1e-12f));
      float* dst = bookemb + (size_t)row * EMB;
#pragma unroll
      for (int j = 0; j < EMB; ++j) dst[j] = rp[j] * inv;
    }
  }
}

// ---------------- kernel: gather + dot scoring (L2-resident tables) ----------------

__global__ __launch_bounds__(256) void score_kernel(
    const int* __restrict__ user_idx, const int* __restrict__ loc_idx,
    const int* __restrict__ pos_idx,  const int* __restrict__ neg_idx,
    const float* __restrict__ user_cat, const float* __restrict__ user_txt,
    const float* __restrict__ loc_cat,  const float* __restrict__ loc_txt,
    const float* __restrict__ bookemb,  float* __restrict__ out)
{
  int b = blockIdx.x * blockDim.x + threadIdx.x;
  if (b >= B_TOTAL) return;
  int ui = user_idx[b], li = loc_idx[b];
  const float4* uc = (const float4*)(user_cat + (size_t)ui * EMB);
  const float4* ut = (const float4*)(user_txt + (size_t)ui * EMB);
  const float4* lc = (const float4*)(loc_cat  + (size_t)li * EMB);
  const float4* lt = (const float4*)(loc_txt  + (size_t)li * EMB);
  const float4* pp = (const float4*)(bookemb + (size_t)pos_idx[b] * EMB);
  const float4* np = (const float4*)(bookemb + (size_t)neg_idx[b] * EMB);
  float ss = 0.f, dp = 0.f, dn = 0.f;
#pragma unroll
  for (int j = 0; j < EMB / 4; ++j) {
    float4 a = uc[j], c = lc[j], t = ut[j], l = lt[j], p = pp[j], q = np[j];
    float u0 = ALPHA_F * (a.x + c.x) + BETA_F * (t.x + l.x);
    float u1 = ALPHA_F * (a.y + c.y) + BETA_F * (t.y + l.y);
    float u2 = ALPHA_F * (a.z + c.z) + BETA_F * (t.z + l.z);
    float u3 = ALPHA_F * (a.w + c.w) + BETA_F * (t.w + l.w);
    ss += u0 * u0 + u1 * u1 + u2 * u2 + u3 * u3;
    dp += u0 * p.x + u1 * p.y + u2 * p.z + u3 * p.w;
    dn += u0 * q.x + u1 * q.y + u2 * q.z + u3 * q.w;
  }
  float inv = rsqrtf(fmaxf(ss, 1e-12f));
  out[2 * b]     = dp * inv * INV_TEMP;
  out[2 * b + 1] = dn * inv * INV_TEMP;
}

// ---------------- launcher ----------------

extern "C" void kernel_launch(void* const* d_in, const int* in_sizes, int n_in,
                              void* d_out, int out_size, void* d_ws, size_t ws_size,
                              hipStream_t stream) {
  const int*   user_idx = (const int*)d_in[0];
  const int*   loc_idx  = (const int*)d_in[1];
  const int*   pos_idx  = (const int*)d_in[2];
  const int*   neg_idx  = (const int*)d_in[3];
  const float* user_cat = (const float*)d_in[4];
  const float* user_txt = (const float*)d_in[5];
  const float* loc_cat  = (const float*)d_in[6];
  const float* loc_txt  = (const float*)d_in[7];
  const float* sbert    = (const float*)d_in[8];
  const float* cat_mh   = (const float*)d_in[9];
  const float* Wc1 = (const float*)d_in[10]; const float* bc1 = (const float*)d_in[11];
  const float* Wc2 = (const float*)d_in[12]; const float* bc2 = (const float*)d_in[13];
  const float* Wt1 = (const float*)d_in[14]; const float* bt1 = (const float*)d_in[15];
  const float* Wt2 = (const float*)d_in[16]; const float* bt2 = (const float*)d_in[17];

  // workspace: [bookemb f32 | Wc1T | Wc2T | Wt1T | Wt2T] (~19.5 MB)
  char* ws = (char*)d_ws;
  float* bookemb = (float*)ws;
  size_t off = (size_t)N_ISBNS * EMB * sizeof(float);
  _Float16* Wc1T = (_Float16*)(ws + off); off += (size_t)H_CAT * NUM_CATS * 2;
  _Float16* Wc2T = (_Float16*)(ws + off); off += (size_t)EMB * H_CAT * 2;
  _Float16* Wt1T = (_Float16*)(ws + off); off += (size_t)H_TXT * SBERT_DIM * 2;
  _Float16* Wt2T = (_Float16*)(ws + off);

  // 1) weights -> f16 [N][K]
  transpose_f16_kernel<<<(NUM_CATS * H_CAT + 255) / 256, 256, 0, stream>>>(Wc1, Wc1T, NUM_CATS, H_CAT);
  transpose_f16_kernel<<<(H_CAT * EMB + 255) / 256, 256, 0, stream>>>(Wc2, Wc2T, H_CAT, EMB);
  transpose_f16_kernel<<<(SBERT_DIM * H_TXT + 255) / 256, 256, 0, stream>>>(Wt1, Wt1T, SBERT_DIM, H_TXT);
  transpose_f16_kernel<<<(H_TXT * EMB + 255) / 256, 256, 0, stream>>>(Wt2, Wt2T, H_TXT, EMB);

  // 2) WMMA-encode all unique books once (25 GFLOP instead of 131 GFLOP)
  encode_books_kernel<<<(N_ISBNS + 63) / 64, 128, 0, stream>>>(
      sbert, cat_mh, Wc1T, bc1, Wc2T, bc2, Wt1T, bt1, Wt2T, bt2, bookemb);

  // 3) bandwidth-bound gather + dot scoring
  score_kernel<<<(B_TOTAL + 255) / 256, 256, 0, stream>>>(
      user_idx, loc_idx, pos_idx, neg_idx,
      user_cat, user_txt, loc_cat, loc_txt, bookemb, (float*)d_out);
}
